// GAT_Surrogate_824633721178
// MI455X (gfx1250) — matrix operand order repaired and checked by reading, not compiled, not run on hardware
//
#include <hip/hip_runtime.h>
#include <hip/hip_bf16.h>

typedef _Float16 h16;
typedef __attribute__((ext_vector_type(16))) _Float16 v16h;
typedef __attribute__((ext_vector_type(8)))  _Float16 v8h;
typedef __attribute__((ext_vector_type(8)))  float    v8f;

#define NEG_SLOPE 0.2f

// ---------------------------------------------------------------------------
// Encoder: h0[n,j] = sum_k x[n,k]*W[k,j] + b[j]   (K=16, memory bound)
// ---------------------------------------------------------------------------
__global__ void k_encoder(const float* __restrict__ x, const float* __restrict__ W,
                          const float* __restrict__ b, float* __restrict__ out, int N) {
    int id = blockIdx.x * blockDim.x + threadIdx.x;
    if (id >= N * 128) return;
    int n = id >> 7, j = id & 127;
    const float* xr = x + (size_t)n * 16;
    float s = b[j];
#pragma unroll
    for (int k = 0; k < 16; ++k) s += xr[k] * W[k * 128 + j];
    out[id] = s;
}

// ---------------------------------------------------------------------------
// f32 -> f16 conversion
// ---------------------------------------------------------------------------
__global__ void k_cvt_f16(const float* __restrict__ in, h16* __restrict__ out, size_t n) {
    size_t id = (size_t)blockIdx.x * blockDim.x + threadIdx.x;
    if (id < n) out[id] = (h16)in[id];
}

// Weight [K,Nout] f32 -> transposed f16 [Nout,K]
__global__ void k_wt_f16(const float* __restrict__ W, h16* __restrict__ Wt, int K, int Nout) {
    int id = blockIdx.x * blockDim.x + threadIdx.x;
    if (id >= K * Nout) return;
    int k = id / Nout, j = id - k * Nout;          // coalesced read of W
    Wt[(size_t)j * K + k] = (h16)W[id];
}

// generic bit-fill
__global__ void k_fill_u32(unsigned* __restrict__ p, unsigned v, size_t n) {
    size_t id = (size_t)blockIdx.x * blockDim.x + threadIdx.x;
    if (id < n) p[id] = v;
}

// ---------------------------------------------------------------------------
// WMMA GEMM: C[M,Nout] = A[M,K](f16) * B (f16, supplied transposed Bt[Nout,K])
// block = 128 threads = 4 waves; block tile = 16 rows x (64*NT) cols;
// wave tile = 16 x (16*NT).  Whole 16xK A panel staged in LDS ONCE (<=16KB),
// so the K loop runs with zero barriers.  All NT B fragments are loaded into
// distinct registers BEFORE the NT WMMAs so the loads overlap in flight.
// Requires M%16==0, K%32==0, Nout%(64*NT)==0 (true for all layers here).
// ---------------------------------------------------------------------------
template <int NT>
__global__ __launch_bounds__(128)
void k_gemm_f16_wmma(const h16* __restrict__ A, const h16* __restrict__ Bt,
                     float* __restrict__ C, int M, int K, int Nout) {
    __shared__ h16 Atile[16][512];                 // supports K up to 512
    const int lane    = threadIdx.x & 31;
    const int wave    = threadIdx.x >> 5;
    const int rowBase = blockIdx.y * 16;
    const int colBase = blockIdx.x * (64 * NT) + wave * (16 * NT);

    // ---- stage full 16 x K A panel (coalesced 8B chunks), one barrier ----
    {
        const int tr = threadIdx.x >> 3;           // 0..15  (row)
        const int tc = (threadIdx.x & 7) * 4;      // 0..28  (col base)
        const h16* arp = A + (size_t)(rowBase + tr) * K;
        for (int c = tc; c < K; c += 32)
            *(uint64_t*)&Atile[tr][c] = *(const uint64_t*)(arp + c);
    }
    __syncthreads();

    const int arow = lane & 15;
    const int ka   = (lane >> 4) * 16;   // A frag: K = ka..ka+15 (contiguous)
    const int kb   = (lane >> 4) * 16;   // B frag: K = kb..kb+15 (contiguous)
    const int bc   = lane & 15;

    const h16* bp[NT];
#pragma unroll
    for (int t = 0; t < NT; ++t)
        bp[t] = Bt + (size_t)(colBase + 16 * t + bc) * K;

    v8f acc[NT] = {};
    for (int k0 = 0; k0 < K; k0 += 32) {
        // issue ALL loads first: 2x ds_load_b128 + NT x (2x global_load_b128)
        union { v16h v; v8h h[2]; } af;
        af.h[0] = *(const v8h*)&Atile[arow][k0 + ka];
        af.h[1] = *(const v8h*)&Atile[arow][k0 + ka + 8];

        v16h bf[NT];
#pragma unroll
        for (int t = 0; t < NT; ++t)
            bf[t] = *(const v16h*)(bp[t] + k0 + kb);

        __builtin_prefetch(bp[0] + k0 + 32 + kb, 0, 3);   // speculative, branchless

        // then the NT back-to-back WMMAs (independent accumulators)
#pragma unroll
        for (int t = 0; t < NT; ++t)
            acc[t] = __builtin_amdgcn_wmma_f32_16x16x32_f16(
                         false, af.v, false, bf[t], (short)0, acc[t], false, false);
    }

    // C layout: lanes 0-15 -> cols, rows 0..7; lanes 16-31 -> rows 8..15
    const int crow0 = rowBase + (lane >> 4) * 8;
#pragma unroll
    for (int t = 0; t < NT; ++t) {
        const int ccol = colBase + 16 * t + (lane & 15);
#pragma unroll
        for (int r = 0; r < 8; ++r)
            C[(size_t)(crow0 + r) * Nout + ccol] = acc[t][r];
    }
}

// ---------------------------------------------------------------------------
// Per-node attention scores: aS[n,h] = sum_c h[n,h,c]*att_src[h,c]  (C=128)
// ---------------------------------------------------------------------------
__global__ __launch_bounds__(128)
void k_attn(const float* __restrict__ Hf, const float* __restrict__ as_,
            const float* __restrict__ ad_, float* __restrict__ aS,
            float* __restrict__ aD, int Fout, int Hh) {
    __shared__ float rs[128], rd[128];
    int n = blockIdx.x, t = threadIdx.x;
    for (int h = 0; h < Hh; ++h) {
        float v = Hf[(size_t)n * Fout + h * 128 + t];
        rs[t] = v * as_[h * 128 + t];
        rd[t] = v * ad_[h * 128 + t];
        __syncthreads();
        for (int s = 64; s > 0; s >>= 1) {
            if (t < s) { rs[t] += rs[t + s]; rd[t] += rd[t + s]; }
            __syncthreads();
        }
        if (t == 0) { aS[n * Hh + h] = rs[0]; aD[n * Hh + h] = rd[0]; }
        __syncthreads();
    }
}

// sign-aware float atomic max via integer atomics (target init = -inf bits)
__device__ __forceinline__ void atomicMaxF(float* addr, float val) {
    if (val >= 0.0f) atomicMax((int*)addr, __float_as_int(val));
    else             atomicMin((unsigned int*)addr, __float_as_uint(val));
}

// pass 1: raw leaky-relu score per edge, segment max over destinations
__global__ void k_edge_max(const int* __restrict__ src, const int* __restrict__ dst,
                           int E, int N, int Hh, const float* __restrict__ aS,
                           const float* __restrict__ aD, float* __restrict__ eAl,
                           float* __restrict__ aM) {
    int e = blockIdx.x * blockDim.x + threadIdx.x;
    int Et = E + N;
    if (e >= Et) return;
    int s = (e < E) ? src[e] : (e - E);
    int d = (e < E) ? dst[e] : (e - E);
    for (int h = 0; h < Hh; ++h) {
        float r = aS[s * Hh + h] + aD[d * Hh + h];
        r = (r > 0.0f) ? r : NEG_SLOPE * r;
        eAl[(size_t)e * Hh + h] = r;
        atomicMaxF(&aM[d * Hh + h], r);
    }
}

// pass 2: exp(raw - max), segment sum of denominators
__global__ void k_edge_sum(const int* __restrict__ src, const int* __restrict__ dst,
                           int E, int N, int Hh, const float* __restrict__ aM,
                           float* __restrict__ eAl, float* __restrict__ aDen) {
    int e = blockIdx.x * blockDim.x + threadIdx.x;
    int Et = E + N;
    if (e >= Et) return;
    int d = (e < E) ? dst[e] : (e - E);
    for (int h = 0; h < Hh; ++h) {
        float a = __expf(eAl[(size_t)e * Hh + h] - aM[d * Hh + h]);
        eAl[(size_t)e * Hh + h] = a;
        atomicAdd(&aDen[d * Hh + h], a);
    }
}

// pass 3: out[d] += (alpha/denom) * h[s]   (coalesced 128-wide feature rows)
__global__ __launch_bounds__(128)
void k_scatter(const int* __restrict__ src, const int* __restrict__ dst,
               int E, int N, int Hh, const float* __restrict__ Hf,
               const float* __restrict__ eAl, const float* __restrict__ aDen,
               float* __restrict__ outacc, int Fout) {
    int e = blockIdx.x, t = threadIdx.x;
    int s = (e < E) ? src[e] : (e - E);
    int d = (e < E) ? dst[e] : (e - E);
    for (int h = 0; h < Hh; ++h) {
        float coef = eAl[(size_t)e * Hh + h] / (aDen[d * Hh + h] + 1e-16f);
        float v = Hf[(size_t)s * Fout + h * 128 + t];
        atomicAdd(&outacc[(size_t)d * Fout + h * 128 + t], coef * v);
    }
}

// bias (+optional relu); layer 3 has H=1 so "mean over heads" is identity
__global__ void k_finalize(float* __restrict__ buf, const float* __restrict__ bias,
                           size_t total, int Fout, int relu) {
    size_t id = (size_t)blockIdx.x * blockDim.x + threadIdx.x;
    if (id >= total) return;
    int j = (int)(id % (size_t)Fout);
    float v = buf[id] + bias[j];
    if (relu) v = fmaxf(v, 0.0f);
    buf[id] = v;
}

// Decoder: out[n, 0..2] = h[n,:] @ dec_W[128,3] + dec_b
__global__ __launch_bounds__(128)
void k_decoder(const float* __restrict__ Hin, const float* __restrict__ W,
               const float* __restrict__ b, float* __restrict__ out, int N) {
    __shared__ float sh[128];
    int n = blockIdx.x, t = threadIdx.x;
    float v = Hin[(size_t)n * 128 + t];
    for (int j = 0; j < 3; ++j) {
        sh[t] = v * W[t * 3 + j];
        __syncthreads();
        for (int s = 64; s > 0; s >>= 1) {
            if (t < s) sh[t] += sh[t + s];
            __syncthreads();
        }
        if (t == 0) out[(size_t)n * 3 + j] = sh[0] + b[j];
        __syncthreads();
    }
}

// ---------------------------------------------------------------------------
// Host-side GAT layer orchestration (all launches on `stream`, capture-safe)
// ---------------------------------------------------------------------------
static void gat_layer(const float* in, int Fin, int Hh,
                      const float* W, const float* as_, const float* ad_,
                      const float* bias, float* outbuf, float* fH,
                      h16* hX, h16* hW,
                      float* aS, float* aD, float* aM, float* aDen, float* eAl,
                      const int* src, const int* dst, int N, int E, int relu,
                      hipStream_t stream) {
    const int Fout = Hh * 128;
    const int Et   = E + N;

    size_t nx = (size_t)N * Fin;
    k_cvt_f16<<<(unsigned)((nx + 255) / 256), 256, 0, stream>>>(in, hX, nx);
    k_wt_f16<<<(unsigned)(((size_t)Fin * Fout + 255) / 256), 256, 0, stream>>>(W, hW, Fin, Fout);

    if (Fout % 256 == 0) {
        dim3 gg(Fout / 256, N / 16);
        k_gemm_f16_wmma<4><<<gg, 128, 0, stream>>>(hX, hW, fH, N, Fin, Fout);
    } else {
        dim3 gg(Fout / 128, N / 16);
        k_gemm_f16_wmma<2><<<gg, 128, 0, stream>>>(hX, hW, fH, N, Fin, Fout);
    }

    k_attn<<<N, 128, 0, stream>>>(fH, as_, ad_, aS, aD, Fout, Hh);

    size_t nh = (size_t)N * Hh;
    k_fill_u32<<<(unsigned)((nh + 255) / 256), 256, 0, stream>>>((unsigned*)aM, 0xFF800000u, nh);   // -inf
    k_fill_u32<<<(unsigned)((nh + 255) / 256), 256, 0, stream>>>((unsigned*)aDen, 0u, nh);
    size_t no = (size_t)N * Fout;
    k_fill_u32<<<(unsigned)((no + 255) / 256), 256, 0, stream>>>((unsigned*)outbuf, 0u, no);

    k_edge_max<<<(Et + 255) / 256, 256, 0, stream>>>(src, dst, E, N, Hh, aS, aD, eAl, aM);
    k_edge_sum<<<(Et + 255) / 256, 256, 0, stream>>>(src, dst, E, N, Hh, aM, eAl, aDen);
    k_scatter<<<Et, 128, 0, stream>>>(src, dst, E, N, Hh, fH, eAl, aDen, outbuf, Fout);

    k_finalize<<<(unsigned)((no + 255) / 256), 256, 0, stream>>>(outbuf, bias, no, Fout, relu);
}

extern "C" void kernel_launch(void* const* d_in, const int* in_sizes, int n_in,
                              void* d_out, int out_size, void* d_ws, size_t ws_size,
                              hipStream_t stream) {
    const int N = in_sizes[0] / 16;     // x: [N,16]
    const int E = in_sizes[1] / 2;      // edge_index: [2,E]

    const float* x     = (const float*)d_in[0];
    const int*   ei    = (const int*)d_in[1];
    const int*   src   = ei;
    const int*   dst   = ei + E;
    const float* enc_W = (const float*)d_in[2];
    const float* enc_b = (const float*)d_in[3];
    const float* g1_W  = (const float*)d_in[4];
    const float* g1_as = (const float*)d_in[5];
    const float* g1_ad = (const float*)d_in[6];
    const float* g1_b  = (const float*)d_in[7];
    const float* g2_W  = (const float*)d_in[8];
    const float* g2_as = (const float*)d_in[9];
    const float* g2_ad = (const float*)d_in[10];
    const float* g2_b  = (const float*)d_in[11];
    const float* g3_W  = (const float*)d_in[12];
    const float* g3_as = (const float*)d_in[13];
    const float* g3_ad = (const float*)d_in[14];
    const float* g3_b  = (const float*)d_in[15];
    const float* dec_W = (const float*)d_in[16];
    const float* dec_b = (const float*)d_in[17];
    float* out = (float*)d_out;
    (void)n_in; (void)out_size; (void)ws_size;

    // workspace carve (256-B aligned blocks)
    char*  w   = (char*)d_ws;
    size_t off = 0;
    auto take = [&](size_t bytes) -> void* {
        void* p = w + off;
        off += (bytes + 255) & ~(size_t)255;
        return p;
    };
    float* fA   = (float*)take((size_t)N * 512 * sizeof(float));   // activation ping
    float* fB   = (float*)take((size_t)N * 512 * sizeof(float));   // activation pong
    float* fH   = (float*)take((size_t)N * 512 * sizeof(float));   // per-layer h (messages)
    h16*   hX   = (h16*)  take((size_t)N * 512 * sizeof(h16));     // f16 copy of GEMM input
    h16*   hW   = (h16*)  take((size_t)512 * 512 * sizeof(h16));   // f16 transposed weight
    float* aS   = (float*)take((size_t)N * 4 * sizeof(float));
    float* aD   = (float*)take((size_t)N * 4 * sizeof(float));
    float* aM   = (float*)take((size_t)N * 4 * sizeof(float));
    float* aDen = (float*)take((size_t)N * 4 * sizeof(float));
    float* eAl  = (float*)take((size_t)(E + N) * 4 * sizeof(float));

    // encoder
    k_encoder<<<(unsigned)(((size_t)N * 128 + 255) / 256), 256, 0, stream>>>(x, enc_W, enc_b, fA, N);

    // three GAT layers
    gat_layer(fA, 128, 4, g1_W, g1_as, g1_ad, g1_b, fB, fH, hX, hW,
              aS, aD, aM, aDen, eAl, src, dst, N, E, /*relu=*/1, stream);
    gat_layer(fB, 512, 4, g2_W, g2_as, g2_ad, g2_b, fA, fH, hX, hW,
              aS, aD, aM, aDen, eAl, src, dst, N, E, /*relu=*/1, stream);
    gat_layer(fA, 512, 1, g3_W, g3_as, g3_ad, g3_b, fB, fH, hX, hW,
              aS, aD, aM, aDen, eAl, src, dst, N, E, /*relu=*/0, stream);

    // decoder
    k_decoder<<<N, 128, 0, stream>>>(fB, dec_W, dec_b, out, N);
}
